// SentenceEncoder_89885075570951
// MI455X (gfx1250) — matrix operand order repaired and checked by reading, not compiled, not run on hardware
//
#include <hip/hip_runtime.h>
#include <hip/hip_bf16.h>

// ---------------------------------------------------------------------------
// SentenceEncoder block for MI455X (gfx1250, wave32, WMMA).
// - All GEMMs: v_wmma_f32_16x16x32_bf16, 64x64 tile per wave (4x4 accums,
//   16 WMMA per K-step, ~32 FLOP/byte of load traffic).
// - Attention: einsum 'nhql,nlhd->nlhd' == colsum_q(softmax) * V, so only the
//   q.k^T energy GEMM uses matrix cores; two-pass branchless online softmax;
//   triangular tile-loop bound (causal mask zeroes everything above q).
// ---------------------------------------------------------------------------

typedef __bf16 bf16_t;
typedef __attribute__((ext_vector_type(16))) __bf16 v16bf;
typedef __attribute__((ext_vector_type(8)))  float  v8f;

constexpr int Bn   = 4;
constexpr int Sn   = 1024;
constexpr int Dn   = 1024;
constexpr int Hn   = 16;
constexpr int DHn  = 64;
constexpr int DFFn = 4096;
constexpr float NEGV = -1.0e20f;

// ---------------------------------------------------------------- f32 -> bf16
__global__ void k_cast_bf16(const float* __restrict__ in, bf16_t* __restrict__ out, int n) {
  int i = blockIdx.x * blockDim.x + threadIdx.x;
  if (i < n) out[i] = (bf16_t)in[i];
}

__global__ void k_zero(float* __restrict__ p, int n) {
  int i = blockIdx.x * blockDim.x + threadIdx.x;
  if (i < n) p[i] = 0.0f;
}

// ------------------------------------------------------------------- GEMM
// C[M,N] = A[M,K] @ W[N,K]^T (+bias, optional ReLU).  One wave per block,
// 64x64 tile = 4x4 WMMA accumulators.  A/B lane layouts per ISA 7.12.2.
template <bool RELU>
__global__ void __launch_bounds__(32)
k_gemm_bf16(const bf16_t* __restrict__ A, const bf16_t* __restrict__ W,
            const float* __restrict__ bias,
            float* __restrict__ C, bf16_t* __restrict__ Cb,
            int M, int N, int K) {
  const int lane = threadIdx.x & 31;
  const int half = lane >> 4;
  const int ln16 = lane & 15;
  const int nBase = blockIdx.x * 64;
  const int mBase = blockIdx.y * 64;

  v8f acc[4][4] = {};

  for (int k = 0; k < K; k += 32) {
    // A tiles: element j <-> K = k + 16*(j>>3) + 8*half + (j&7)
    v16bf a[4];
#pragma unroll
    for (int mt = 0; mt < 4; ++mt) {
      const bf16_t* ap = A + (size_t)(mBase + mt * 16 + ln16) * K + k + 8 * half;
      union { v16bf v; uint4 u[2]; } ua;
      ua.u[0] = *(const uint4*)(ap);
      ua.u[1] = *(const uint4*)(ap + 16);
      a[mt] = ua.v;
    }
    // B tiles: lane = column n, element j <-> K = k + 16*(j>>3) + 8*half + (j&7)
    v16bf bm[4];
#pragma unroll
    for (int nt = 0; nt < 4; ++nt) {
      const bf16_t* bp = W + (size_t)(nBase + nt * 16 + ln16) * K + k + 8 * half;
      union { v16bf v; uint4 u[2]; } ub;
      ub.u[0] = *(const uint4*)(bp);
      ub.u[1] = *(const uint4*)(bp + 16);
      bm[nt] = ub.v;
    }
    if (k + 32 < K) {
      __builtin_prefetch(A + (size_t)(mBase + ln16) * K + k + 32, 0, 1);
      __builtin_prefetch(W + (size_t)(nBase + ln16) * K + k + 32, 0, 1);
    }
#pragma unroll
    for (int mt = 0; mt < 4; ++mt)
#pragma unroll
      for (int nt = 0; nt < 4; ++nt)
        acc[mt][nt] = __builtin_amdgcn_wmma_f32_16x16x32_bf16(
            false, a[mt], false, bm[nt], (short)0, acc[mt][nt], false, false);
  }

#pragma unroll
  for (int mt = 0; mt < 4; ++mt)
#pragma unroll
    for (int nt = 0; nt < 4; ++nt) {
      int col = nBase + nt * 16 + ln16;
      float bv = bias ? bias[col] : 0.0f;
#pragma unroll
      for (int r = 0; r < 8; ++r) {
        int row = mBase + mt * 16 + r + 8 * half;
        float v = acc[mt][nt][r] + bv;
        if (RELU) v = fmaxf(v, 0.0f);
        size_t idx = (size_t)row * N + col;
        if (C)  C[idx]  = v;
        if (Cb) Cb[idx] = (bf16_t)v;
      }
    }
}

// ------------------------------------------------- layernorm over 64 (per head)
__global__ void k_ln64(const float* __restrict__ in, const float* __restrict__ g,
                       const float* __restrict__ b, float* __restrict__ out,
                       bf16_t* __restrict__ outb, int nrows) {
  int gid = blockIdx.x * blockDim.x + threadIdx.x;
  int wave = gid >> 5, lane = gid & 31;
  if (wave >= nrows) return;
  float2 x2 = *(const float2*)(in + (size_t)wave * 64 + lane * 2);
  float s = x2.x + x2.y;
#pragma unroll
  for (int off = 16; off >= 1; off >>= 1) s += __shfl_xor(s, off, 32);
  float mu = s * (1.0f / 64.0f);
  float d0 = x2.x - mu, d1 = x2.y - mu;
  float vv = d0 * d0 + d1 * d1;
#pragma unroll
  for (int off = 16; off >= 1; off >>= 1) vv += __shfl_xor(vv, off, 32);
  float rstd = rsqrtf(vv * (1.0f / 64.0f) + 1e-5f);
  float y0 = d0 * rstd * g[lane * 2]     + b[lane * 2];
  float y1 = d1 * rstd * g[lane * 2 + 1] + b[lane * 2 + 1];
  size_t o = (size_t)wave * 64 + lane * 2;
  if (out)  { out[o] = y0;  out[o + 1] = y1; }
  if (outb) { outb[o] = (bf16_t)y0; outb[o + 1] = (bf16_t)y1; }
}

// ------------------------------------------- layernorm over 1024 (+ residual)
__global__ void __launch_bounds__(256)
k_ln1024(const float* __restrict__ in, const float* __restrict__ res,
         const float* __restrict__ g, const float* __restrict__ b,
         float* __restrict__ out, bf16_t* __restrict__ outb) {
  const int row = blockIdx.x;
  const int t = threadIdx.x;
  const float* p = in + (size_t)row * 1024;
  const float* rp = res ? res + (size_t)row * 1024 : nullptr;
  __shared__ float red[8];
  float x[4];
  float loc = 0.0f;
#pragma unroll
  for (int i = 0; i < 4; ++i) {
    int c = t + i * 256;
    x[i] = p[c] + (rp ? rp[c] : 0.0f);
    loc += x[i];
  }
  float w = loc;
#pragma unroll
  for (int off = 16; off >= 1; off >>= 1) w += __shfl_xor(w, off, 32);
  if ((t & 31) == 0) red[t >> 5] = w;
  __syncthreads();
  float mu = 0.0f;
#pragma unroll
  for (int i = 0; i < 8; ++i) mu += red[i];
  mu *= (1.0f / 1024.0f);
  __syncthreads();
  float vloc = 0.0f;
#pragma unroll
  for (int i = 0; i < 4; ++i) { float d = x[i] - mu; vloc += d * d; }
  w = vloc;
#pragma unroll
  for (int off = 16; off >= 1; off >>= 1) w += __shfl_xor(w, off, 32);
  if ((t & 31) == 0) red[t >> 5] = w;
  __syncthreads();
  float var = 0.0f;
#pragma unroll
  for (int i = 0; i < 8; ++i) var += red[i];
  float rstd = rsqrtf(var * (1.0f / 1024.0f) + 1e-5f);
#pragma unroll
  for (int i = 0; i < 4; ++i) {
    int c = t + i * 256;
    float y = (x[i] - mu) * rstd * g[c] + b[c];
    size_t idx = (size_t)row * 1024 + c;
    if (out)  out[idx]  = y;
    if (outb) outb[idx] = (bf16_t)y;
  }
}

// -------------------------------------------------- attention column-sums
// One wave per (b, h, 16 q-rows).  Pass 1: WMMA energy + BRANCHLESS online
// max/sum (no EXEC divergence around WMMA).  Pass 2: recompute exp/Z and
// atomicAdd column sums.  Causal mask => only tiles lt <= blockIdx.x matter
// (block 0 sweeps all tiles so the fully-masked q=0 row stays exact).
__global__ void __launch_bounds__(32)
k_attn_colsum(const bf16_t* __restrict__ qb, const bf16_t* __restrict__ kb,
              const float* __restrict__ mask, float* __restrict__ colsum) {
  const int lane = threadIdx.x & 31;
  const int half = lane >> 4;
  const int ln16 = lane & 15;
  const int qBase = blockIdx.x * 16;
  const int h = blockIdx.y;
  const int b = blockIdx.z;
  const int nTiles = (blockIdx.x == 0) ? (Sn / 16) : (blockIdx.x + 1);

  v16bf aq[2];
#pragma unroll
  for (int ks = 0; ks < 2; ++ks) {
    const bf16_t* ap = qb + ((size_t)(b * Sn + qBase + ln16)) * Dn + h * DHn + ks * 32 + 8 * half;
    union { v16bf v; uint4 u[2]; } ua;
    ua.u[0] = *(const uint4*)(ap);
    ua.u[1] = *(const uint4*)(ap + 16);
    aq[ks] = ua.v;
  }
  float mq[8];
#pragma unroll
  for (int r = 0; r < 8; ++r) mq[r] = mask[b * Sn + qBase + r + 8 * half];

  const float inv = 1.0f / 32.0f;  // 1/sqrt(D)
  float lm[8], ls[8];
#pragma unroll
  for (int r = 0; r < 8; ++r) { lm[r] = -3.0e38f; ls[r] = 0.0f; }

  for (int lt = 0; lt < nTiles; ++lt) {
    int l = lt * 16 + ln16;
    const bf16_t* bp = kb + ((size_t)(b * Sn + l)) * Dn + h * DHn + 8 * half;
    union { v16bf v; uint4 u[2]; } ub0, ub1;
    ub0.u[0] = *(const uint4*)(bp);      ub0.u[1] = *(const uint4*)(bp + 16);
    ub1.u[0] = *(const uint4*)(bp + 32); ub1.u[1] = *(const uint4*)(bp + 48);
    v8f sc = {};
    sc = __builtin_amdgcn_wmma_f32_16x16x32_bf16(false, aq[0], false, ub0.v, (short)0, sc, false, false);
    sc = __builtin_amdgcn_wmma_f32_16x16x32_bf16(false, aq[1], false, ub1.v, (short)0, sc, false, false);
    float ml = mask[b * Sn + l];
#pragma unroll
    for (int r = 0; r < 8; ++r) {
      int q = qBase + r + 8 * half;
      float dm = (q > l) ? 0.0f : NEGV;
      float pre = sc[r] * inv + mq[r] * ml * dm - fabsf((float)(q - l));
      float nm = fmaxf(lm[r], pre);                       // branchless online update
      ls[r] = ls[r] * __expf(lm[r] - nm) + __expf(pre - nm);
      lm[r] = nm;
    }
  }
  // merge stats across the 16 lanes of each half-group
  float gm[8], gz[8];
#pragma unroll
  for (int r = 0; r < 8; ++r) {
    float m = lm[r], s = ls[r];
#pragma unroll
    for (int off = 8; off >= 1; off >>= 1) {
      float om = __shfl_xor(m, off, 32);
      float os = __shfl_xor(s, off, 32);
      float nm = fmaxf(m, om);
      s = s * __expf(m - nm) + os * __expf(om - nm);
      m = nm;
    }
    gm[r] = m;
    gz[r] = (s > 0.0f) ? 1.0f / s : 0.0f;
  }
  // pass 2: column sums of normalized attention
  for (int lt = 0; lt < nTiles; ++lt) {
    int l = lt * 16 + ln16;
    const bf16_t* bp = kb + ((size_t)(b * Sn + l)) * Dn + h * DHn + 8 * half;
    union { v16bf v; uint4 u[2]; } ub0, ub1;
    ub0.u[0] = *(const uint4*)(bp);      ub0.u[1] = *(const uint4*)(bp + 16);
    ub1.u[0] = *(const uint4*)(bp + 32); ub1.u[1] = *(const uint4*)(bp + 48);
    v8f sc = {};
    sc = __builtin_amdgcn_wmma_f32_16x16x32_bf16(false, aq[0], false, ub0.v, (short)0, sc, false, false);
    sc = __builtin_amdgcn_wmma_f32_16x16x32_bf16(false, aq[1], false, ub1.v, (short)0, sc, false, false);
    float ml = mask[b * Sn + l];
    float cs = 0.0f;
#pragma unroll
    for (int r = 0; r < 8; ++r) {
      int q = qBase + r + 8 * half;
      float dm = (q > l) ? 0.0f : NEGV;
      float pre = sc[r] * inv + mq[r] * ml * dm - fabsf((float)(q - l));
      cs += __expf(pre - gm[r]) * gz[r];
    }
    cs += __shfl_xor(cs, 16, 32);
    if (half == 0)
      atomicAdd(colsum + ((size_t)(b * Hn + h)) * Sn + l, cs);
  }
}

// ----------------------------------- out[l,h,d] = colsum[b,h,l] * vn[l,h,d]
__global__ void k_scale_v(const float* __restrict__ vn, const float* __restrict__ colsum,
                          bf16_t* __restrict__ outb, int total) {
  int i = blockIdx.x * blockDim.x + threadIdx.x;
  if (i >= total) return;
  int d = i & (Dn - 1);
  int row = i >> 10;          // /Dn
  int h = d >> 6;             // /DHn
  int l = row & (Sn - 1);
  int b = row >> 10;          // /Sn
  float v = vn[i] * colsum[((size_t)(b * Hn + h)) * Sn + l];
  outb[i] = (bf16_t)v;
}

// ------------------------------------- fusion gate: preg = f*sf + (1-f)*hf
__global__ void k_gate(const float* __restrict__ sf, const float* __restrict__ hf,
                       const float* __restrict__ bf_, float* __restrict__ preg, int total) {
  int i = blockIdx.x * blockDim.x + threadIdx.x;
  if (i >= total) return;
  int col = i & (Dn - 1);
  float s = sf[i], h = hf[i];
  float f = 1.0f / (1.0f + __expf(-(s + h + bf_[col])));
  preg[i] = f * s + (1.0f - f) * h;
}

// ---------------------------------------------------------------------------
extern "C" void kernel_launch(void* const* d_in, const int* in_sizes, int n_in,
                              void* d_out, int out_size, void* d_ws, size_t ws_size,
                              hipStream_t stream) {
  (void)in_sizes; (void)n_in; (void)out_size; (void)ws_size;
  const float* x      = (const float*)d_in[0];
  const float* mask   = (const float*)d_in[1];
  const float* Wq     = (const float*)d_in[2];
  const float* Wk     = (const float*)d_in[3];
  const float* Wv     = (const float*)d_in[4];
  const float* ln1_g  = (const float*)d_in[5];
  const float* ln1_b  = (const float*)d_in[6];
  const float* Wo     = (const float*)d_in[7];
  const float* bo     = (const float*)d_in[8];
  const float* ln2_g  = (const float*)d_in[9];
  const float* ln2_b  = (const float*)d_in[10];
  const float* Ws     = (const float*)d_in[11];
  const float* Wh     = (const float*)d_in[12];
  const float* b_f    = (const float*)d_in[13];
  const float* lnfg_g = (const float*)d_in[14];
  const float* lnfg_b = (const float*)d_in[15];
  const float* Wp1    = (const float*)d_in[16];
  const float* bp1    = (const float*)d_in[17];
  const float* Wp2    = (const float*)d_in[18];
  const float* bp2    = (const float*)d_in[19];
  const float* lnffn_g = (const float*)d_in[20];
  const float* lnffn_b = (const float*)d_in[21];
  float* out = (float*)d_out;

  const int M = Bn * Sn;  // 4096
  char* ws = (char*)d_ws;
  size_t off = 0;
  auto alloc = [&](size_t bytes) -> void* {
    void* p = ws + off;
    off += (bytes + 255) & ~(size_t)255;
    return p;
  };

  bf16_t* xb   = (bf16_t*)alloc((size_t)M * Dn * 2);
  bf16_t* Wqb  = (bf16_t*)alloc((size_t)Dn * Dn * 2);
  bf16_t* Wkb  = (bf16_t*)alloc((size_t)Dn * Dn * 2);
  bf16_t* Wvb  = (bf16_t*)alloc((size_t)Dn * Dn * 2);
  bf16_t* Wob  = (bf16_t*)alloc((size_t)Dn * Dn * 2);
  bf16_t* Wsb  = (bf16_t*)alloc((size_t)Dn * Dn * 2);
  bf16_t* Whb  = (bf16_t*)alloc((size_t)Dn * Dn * 2);
  bf16_t* Wp1b = (bf16_t*)alloc((size_t)DFFn * Dn * 2);
  bf16_t* Wp2b = (bf16_t*)alloc((size_t)Dn * DFFn * 2);
  float*  qf   = (float*)alloc((size_t)M * Dn * 4);
  float*  kf   = (float*)alloc((size_t)M * Dn * 4);
  float*  vf   = (float*)alloc((size_t)M * Dn * 4);
  bf16_t* qnb  = (bf16_t*)alloc((size_t)M * Dn * 2);
  bf16_t* knb  = (bf16_t*)alloc((size_t)M * Dn * 2);
  float*  vn   = (float*)alloc((size_t)M * Dn * 4);
  float*  colsum = (float*)alloc((size_t)Bn * Hn * Sn * 4);
  bf16_t* ainb = (bf16_t*)alloc((size_t)M * Dn * 2);
  float*  ao   = (float*)alloc((size_t)M * Dn * 4);
  float*  hb   = (float*)alloc((size_t)M * Dn * 4);
  bf16_t* hbb  = (bf16_t*)alloc((size_t)M * Dn * 2);
  float*  tmp1 = (float*)alloc((size_t)M * Dn * 4);
  float*  tmp2 = (float*)alloc((size_t)M * Dn * 4);
  float*  sfb  = (float*)alloc((size_t)M * Dn * 4);
  float*  hfb  = (float*)alloc((size_t)M * Dn * 4);
  float*  preg = (float*)alloc((size_t)M * Dn * 4);
  float*  gf   = (float*)alloc((size_t)M * Dn * 4);
  bf16_t* gb   = (bf16_t*)alloc((size_t)M * Dn * 2);
  bf16_t* ff1b = (bf16_t*)alloc((size_t)M * DFFn * 2);
  float*  ff2  = (float*)alloc((size_t)M * Dn * 4);

  auto cast = [&](const float* src, bf16_t* dst, int n) {
    k_cast_bf16<<<(n + 255) / 256, 256, 0, stream>>>(src, dst, n);
  };
  cast(x, xb, M * Dn);
  cast(Wq, Wqb, Dn * Dn);  cast(Wk, Wkb, Dn * Dn);  cast(Wv, Wvb, Dn * Dn);
  cast(Wo, Wob, Dn * Dn);  cast(Ws, Wsb, Dn * Dn);  cast(Wh, Whb, Dn * Dn);
  cast(Wp1, Wp1b, DFFn * Dn);  cast(Wp2, Wp2b, Dn * DFFn);

  auto gemm = [&](const bf16_t* A, const bf16_t* W, const float* bias,
                  float* C, bf16_t* Cb, int Mm, int Nn, int Kk, bool relu) {
    dim3 grid(Nn / 64, Mm / 64);
    if (relu) k_gemm_bf16<true><<<grid, 32, 0, stream>>>(A, W, bias, C, Cb, Mm, Nn, Kk);
    else      k_gemm_bf16<false><<<grid, 32, 0, stream>>>(A, W, bias, C, Cb, Mm, Nn, Kk);
  };

  // QKV projections
  gemm(xb, Wqb, nullptr, qf, nullptr, M, Dn, Dn, false);
  gemm(xb, Wkb, nullptr, kf, nullptr, M, Dn, Dn, false);
  gemm(xb, Wvb, nullptr, vf, nullptr, M, Dn, Dn, false);

  // per-head layernorm (rows of 64 are contiguous since D = H*DH)
  int headrows = M * Hn;
  k_ln64<<<(headrows * 32 + 255) / 256, 256, 0, stream>>>(qf, ln1_g, ln1_b, nullptr, qnb, headrows);
  k_ln64<<<(headrows * 32 + 255) / 256, 256, 0, stream>>>(kf, ln1_g, ln1_b, nullptr, knb, headrows);
  k_ln64<<<(headrows * 32 + 255) / 256, 256, 0, stream>>>(vf, ln1_g, ln1_b, vn, nullptr, headrows);

  // attention column sums (softmax over keys, summed over queries)
  k_zero<<<(Bn * Hn * Sn + 255) / 256, 256, 0, stream>>>(colsum, Bn * Hn * Sn);
  k_attn_colsum<<<dim3(Sn / 16, Hn, Bn), 32, 0, stream>>>(qnb, knb, mask, colsum);
  k_scale_v<<<(M * Dn + 255) / 256, 256, 0, stream>>>(vn, colsum, ainb, M * Dn);

  // output projection + ln2
  gemm(ainb, Wob, bo, ao, nullptr, M, Dn, Dn, false);
  k_ln1024<<<M, 256, 0, stream>>>(ao, nullptr, ln2_g, ln2_b, hb, hbb);

  // fusion gate branches
  gemm(xb, Wsb, nullptr, tmp1, nullptr, M, Dn, Dn, false);
  gemm(hbb, Whb, nullptr, tmp2, nullptr, M, Dn, Dn, false);
  k_ln1024<<<M, 256, 0, stream>>>(tmp1, nullptr, lnfg_g, lnfg_b, sfb, nullptr);
  k_ln1024<<<M, 256, 0, stream>>>(tmp2, nullptr, lnfg_g, lnfg_b, hfb, nullptr);
  k_gate<<<(M * Dn + 255) / 256, 256, 0, stream>>>(sfb, hfb, b_f, preg, M * Dn);
  k_ln1024<<<M, 256, 0, stream>>>(preg, nullptr, lnfg_g, lnfg_b, gf, gb);

  // FFN + residual + final layernorm
  gemm(gb, Wp1b, bp1, nullptr, ff1b, M, DFFn, Dn, true);
  gemm(ff1b, Wp2b, bp2, ff2, nullptr, M, Dn, DFFn, false);
  k_ln1024<<<M, 256, 0, stream>>>(ff2, gf, lnffn_g, lnffn_b, out, nullptr);
}
